// AgentTokenBlock_6141803233371
// MI455X (gfx1250) — matrix operand order repaired and checked by reading, not compiled, not run on hardware
//
#include <hip/hip_runtime.h>

// ---------------------------------------------------------------------------
// MI455X (gfx1250) implementation of AgentTokenBlock.
// wave32, WMMA f32_16x16x32_bf16 for all heavy matmuls, fused cross-attn so
// the 536MB z_tokens tensor is streamed from HBM exactly once.
// Round 4: contiguous global->LDS staging moved to CDNA5 async-DMA
// (global_load_async_to_lds_b128 / s_wait_asynccnt); transpose staging uses
// independent registers so loads pipeline instead of serializing.
// ---------------------------------------------------------------------------

typedef unsigned short ushort_t;
typedef __attribute__((ext_vector_type(16))) __bf16 v16bf;
typedef __attribute__((ext_vector_type(8)))  float  v8f;

union FragU { v16bf v; unsigned int u[8]; uint4 q[2]; };
union PackU { uint4 q; ushort_t s[8]; };

static constexpr int Bb = 2, Tt = 256, Ss = 256, Dd = 1024;
static constexpr int Hh = 16, Kk = 4, HD = 64, Gg = 4, HF = 4096;

__device__ __forceinline__ ushort_t f2bf(float f) {
    unsigned u = __float_as_uint(f);
    unsigned r = ((u >> 16) & 1u) + 0x7fffu;
    return (ushort_t)((u + r) >> 16);
}
__device__ __forceinline__ float bf2f(ushort_t s) {
    return __uint_as_float(((unsigned)s) << 16);
}
__device__ __forceinline__ v8f vzero8() {
    v8f v;
#pragma unroll
    for (int i = 0; i < 8; i++) v[i] = 0.f;
    return v;
}
__device__ __forceinline__ v8f wmma_bf16(v16bf a, v16bf b, v8f c) {
    // D = A(16x32) * B(32x16) + C, fp32 accumulate
    return __builtin_amdgcn_wmma_f32_16x16x32_bf16(false, a, false, b, (short)0, c, false, false);
}
// CDNA5 async global->LDS copy (ASYNCcnt-tracked DMA, no VGPR round trip).
__device__ __forceinline__ void async_copy_b128(const void* gptr, void* lptr) {
    unsigned lds = (unsigned)(uintptr_t)lptr;              // LDS offset = addr[31:0]
    unsigned long long ga = (unsigned long long)(uintptr_t)gptr;
    asm volatile("global_load_async_to_lds_b128 %0, %1, off" :: "v"(lds), "v"(ga) : "memory");
}
__device__ __forceinline__ void async_wait0() {
    asm volatile("s_wait_asynccnt 0x0" ::: "memory");
}
__device__ __forceinline__ float wred_sum(float v) {   // full wave32
    for (int m = 16; m >= 1; m >>= 1) v += __shfl_xor(v, m, 32);
    return v;
}
__device__ __forceinline__ float hred_sum(float v) {   // within 16-lane half
    for (int m = 1; m < 16; m <<= 1) v += __shfl_xor(v, m, 32);
    return v;
}
__device__ __forceinline__ float hred_max(float v) {
    for (int m = 1; m < 16; m <<= 1) v = fmaxf(v, __shfl_xor(v, m, 32));
    return v;
}

// A-fragment (16x32 bf16) from row-major bf16 tile with row stride `stride`.
__device__ __forceinline__ void load_afrag(FragU& f, const ushort_t* base, int stride, int lane) {
    int r  = lane & 15;
    int kb = (lane < 16) ? 0 : 16;
    const ushort_t* ar = base + (size_t)r * stride;
#pragma unroll
    for (int p = 0; p < 4; p++) f.u[p]     = *(const unsigned int*)(ar + kb + 2 * p);
#pragma unroll
    for (int p = 0; p < 4; p++) f.u[4 + p] = *(const unsigned int*)(ar + 8 + kb + 2 * p);
}

// ---------------------------------------------------------------------------
// fp32 -> bf16 conversion (vectorized: float4 in, 4x bf16 out). n % 4 == 0.
// ---------------------------------------------------------------------------
__global__ void k_f32_to_bf16(const float* __restrict__ src, ushort_t* __restrict__ dst, int n) {
    int i = blockIdx.x * blockDim.x + threadIdx.x;
    int stride = gridDim.x * blockDim.x;
    int n4 = n >> 2;
    for (; i < n4; i += stride) {
        float4 f = ((const float4*)src)[i];
        union { uint2 u; ushort_t s[4]; } o;
        o.s[0] = f2bf(f.x); o.s[1] = f2bf(f.y); o.s[2] = f2bf(f.z); o.s[3] = f2bf(f.w);
        ((uint2*)dst)[i] = o.u;
    }
}

// ---------------------------------------------------------------------------
// Transpose + convert: dst[n][k] = bf16(src[k][n]); src is K x N fp32.
// ---------------------------------------------------------------------------
__global__ __launch_bounds__(256) void k_transpose_bf16(const float* __restrict__ src,
                                                        ushort_t* __restrict__ dst,
                                                        int K, int N) {
    __shared__ float tile[32][33];
    int k0 = blockIdx.y * 32, n0 = blockIdx.x * 32;
    int tx = threadIdx.x & 31, ty = threadIdx.x >> 5;     // 8 rows per pass
#pragma unroll
    for (int i = 0; i < 4; i++)
        tile[ty + i * 8][tx] = src[(size_t)(k0 + ty + i * 8) * N + n0 + tx];
    __syncthreads();
#pragma unroll
    for (int i = 0; i < 4; i++)
        dst[(size_t)(n0 + ty + i * 8) * K + k0 + tx] = f2bf(tile[tx][ty + i * 8]);
}

// ---------------------------------------------------------------------------
// Row RMSNorm (D=1024) -> bf16
// ---------------------------------------------------------------------------
__global__ __launch_bounds__(256) void k_rmsnorm_bf16(const float* __restrict__ x,
                                                      const float* __restrict__ g,
                                                      ushort_t* __restrict__ out, int D) {
    int row = blockIdx.x;
    const float* xr = x + (size_t)row * D;
    __shared__ float red8[8];
    float ss = 0.f;
    for (int d = threadIdx.x; d < D; d += 256) { float v = xr[d]; ss += v * v; }
    ss = wred_sum(ss);
    if ((threadIdx.x & 31) == 0) red8[threadIdx.x >> 5] = ss;
    __syncthreads();
    float tot = 0.f;
#pragma unroll
    for (int i = 0; i < 8; i++) tot += red8[i];
    float inv = rsqrtf(tot / (float)D + 1e-6f);
    for (int d = threadIdx.x; d < D; d += 256) out[(size_t)row * D + d] = f2bf(xr[d] * inv * g[d]);
}

// ---------------------------------------------------------------------------
// Per-64 head RMSNorm (gain != null) or plain convert -> bf16
// ---------------------------------------------------------------------------
__global__ __launch_bounds__(256) void k_headnorm_bf16(const float* __restrict__ x,
                                                       const float* __restrict__ gain,
                                                       ushort_t* __restrict__ out, int C) {
    int row = blockIdx.x;
    const float* xr = x + (size_t)row * C;
    __shared__ float gss[16];
    if (threadIdx.x < 16) gss[threadIdx.x] = 0.f;
    __syncthreads();
    for (int d = threadIdx.x; d < C; d += 256) { float v = xr[d]; atomicAdd(&gss[d >> 6], v * v); }
    __syncthreads();
    for (int d = threadIdx.x; d < C; d += 256) {
        float v = xr[d];
        if (gain) v = v * rsqrtf(gss[d >> 6] * (1.f / 64.f) + 1e-6f) * gain[d & 63];
        out[(size_t)row * C + d] = f2bf(v);
    }
}

// ---------------------------------------------------------------------------
// Generic bf16 WMMA GEMM: C[MxN](f32) = A[MxK]bf16 @ B[KxN]bf16 (+residual)
// Block tile 128x128, 8 waves (4m x 2n), each wave 32x64 = 2x4 WMMA accs.
// A tile staged via async DMA; B tile register-staged + transposed.
// lda/ldb multiples of 8, M%128==0, N%128==0.
// ---------------------------------------------------------------------------
__global__ __launch_bounds__(256) void k_gemm_bf16(const ushort_t* __restrict__ A,
                                                   const ushort_t* __restrict__ B,
                                                   const float* __restrict__ Cres,
                                                   float* __restrict__ Cout,
                                                   ushort_t* __restrict__ CoutBf,
                                                   int M, int N, int Kd,
                                                   int lda, int ldb, int ldc) {
    __shared__ __align__(16) ushort_t As[128 * 40];   // padded stride 40 (=80B, 16B aligned)
    __shared__ __align__(16) ushort_t Bt[128 * 40];   // transposed: Bt[n][k]
    int m0 = blockIdx.y * 128, n0 = blockIdx.x * 128;
    int tid = threadIdx.x, lane = tid & 31, w = tid >> 5;
    int wm = (w >> 1) * 32, wn = (w & 1) * 64;
    v8f acc[2][4];
#pragma unroll
    for (int mi = 0; mi < 2; mi++)
#pragma unroll
        for (int ni = 0; ni < 4; ni++) acc[mi][ni] = vzero8();

    for (int k0 = 0; k0 < Kd; k0 += 32) {
        __syncthreads();
        // stage A: 128x32 = 512 b128 chunks, 2 per thread, async DMA to LDS
#pragma unroll
        for (int i = 0; i < 2; i++) {
            int idx = tid + 256 * i;
            int r = idx >> 2, c = (idx & 3) * 8;
            async_copy_b128(&A[(size_t)(m0 + r) * lda + k0 + c], &As[r * 40 + c]);
        }
        // stage B: 32x128 = 512 b128 chunks, 2 per thread (independent regs),
        // scatter-transpose into Bt
        PackU tb[2]; int rB[2], cB[2];
#pragma unroll
        for (int i = 0; i < 2; i++) {
            int idx = tid + 256 * i;
            rB[i] = idx >> 4; cB[i] = (idx & 15) * 8;
            tb[i].q = *(const uint4*)&B[(size_t)(k0 + rB[i]) * ldb + n0 + cB[i]];
        }
#pragma unroll
        for (int i = 0; i < 2; i++)
#pragma unroll
            for (int e = 0; e < 8; e++) Bt[(cB[i] + e) * 40 + rB[i]] = tb[i].s[e];
        if (k0 + 32 < Kd) {                          // prefetch next tiles, keep near
            if (tid < 128) __builtin_prefetch(&A[(size_t)(m0 + tid) * lda + k0 + 32], 0, 3);
            if (tid < 64) {
                int r = tid & 31, c = (tid >> 5) * 64;
                __builtin_prefetch(&B[(size_t)(k0 + 32 + r) * ldb + n0 + c], 0, 3);
            }
        }
        async_wait0();
        __syncthreads();

        FragU fa[2], fb[4];
#pragma unroll
        for (int mi = 0; mi < 2; mi++)
            load_afrag(fa[mi], &As[(wm + mi * 16) * 40], 40, lane);
        int koff = (lane < 16) ? 0 : 16;
#pragma unroll
        for (int ni = 0; ni < 4; ni++) {
            int n = wn + ni * 16 + (lane & 15);
            const ushort_t* vr = &Bt[n * 40 + koff];
            fb[ni].q[0] = *(const uint4*)vr;
            fb[ni].q[1] = *(const uint4*)(vr + 8);
        }
#pragma unroll
        for (int mi = 0; mi < 2; mi++)
#pragma unroll
            for (int ni = 0; ni < 4; ni++)
                acc[mi][ni] = wmma_bf16(fa[mi].v, fb[ni].v, acc[mi][ni]);
    }

#pragma unroll
    for (int mi = 0; mi < 2; mi++)
#pragma unroll
        for (int ni = 0; ni < 4; ni++)
#pragma unroll
            for (int i = 0; i < 8; i++) {
                int row = m0 + wm + mi * 16 + i + ((lane < 16) ? 0 : 8);
                int col = n0 + wn + ni * 16 + (lane & 15);
                float v = acc[mi][ni][i];
                if (Cres)   v += Cres[(size_t)row * ldc + col];
                if (Cout)   Cout[(size_t)row * ldc + col] = v;
                if (CoutBf) CoutBf[(size_t)row * ldc + col] = f2bf(v);
            }
}

// ---------------------------------------------------------------------------
// Fused cross-attention: one block per (b,t). Streams z once, WMMA K/V proj
// against pre-transposed bf16 weights, online softmax with tanh softcap,
// output projection + residual.
// ---------------------------------------------------------------------------
__global__ __launch_bounds__(256) void k_cross(const float* __restrict__ agent,
                                               const float* __restrict__ z,
                                               const float* __restrict__ norm1_g,
                                               const float* __restrict__ normkv_g,
                                               const float* __restrict__ c_wq,
                                               const ushort_t* __restrict__ wkT,   // [256][1024] bf16
                                               const ushort_t* __restrict__ wvT,   // [256][1024] bf16
                                               const float* __restrict__ c_wo,
                                               const float* __restrict__ c_qg,
                                               const float* __restrict__ c_kg,
                                               float* __restrict__ agent1) {
    int bt = blockIdx.x;                                  // b*T + t
    const float* arow  = agent + (size_t)bt * Dd;
    const float* zbase = z + (size_t)bt * Ss * Dd;

    __shared__ __align__(16) ushort_t a_bf[Dd];
    __shared__ __align__(16) ushort_t qn_bf[Hh * HD];
    __shared__ __align__(16) ushort_t z_bf[16 * 1032];    // 16 rows, padded stride
    __shared__ __align__(16) ushort_t kv_bf[16 * 520];    // cols 0..255 = k(normed), 256..511 = v
    __shared__ float sc_t[Hh * 16];
    __shared__ float red8[8];
    __shared__ float qss[Hh];
    __shared__ float mrun[Hh], lrun[Hh], alpha[Hh];
    __shared__ float o_sh[Hh * HD];

    int tid = threadIdx.x, lane = tid & 31, w = tid >> 5;
    int h = tid >> 4, x16 = tid & 15;

    // ---- 1. RMSNorm agent row ----
    float av[4]; float ss = 0.f;
#pragma unroll
    for (int i = 0; i < 4; i++) { av[i] = arow[tid + 256 * i]; ss += av[i] * av[i]; }
    ss = wred_sum(ss);
    if (lane == 0) red8[w] = ss;
    if (tid < Hh) { qss[tid] = 0.f; mrun[tid] = -1e30f; lrun[tid] = 0.f; }
    __syncthreads();
    float tot = 0.f;
#pragma unroll
    for (int i = 0; i < 8; i++) tot += red8[i];
    float inv = rsqrtf(tot * (1.f / Dd) + 1e-6f);
#pragma unroll
    for (int i = 0; i < 4; i++) { int d = tid + 256 * i; a_bf[d] = f2bf(av[i] * inv * norm1_g[d]); }
    __syncthreads();

    // ---- 2. q = a_norm @ c_wq (GEMV, 4 coalesced output streams/thread) ----
    float qv[4] = {0.f, 0.f, 0.f, 0.f};
    for (int d = 0; d < Dd; d++) {
        float aval = bf2f(a_bf[d]);
        const float* wr = c_wq + (size_t)d * (Hh * HD);
#pragma unroll
        for (int i = 0; i < 4; i++) qv[i] += aval * wr[tid + 256 * i];
    }
#pragma unroll
    for (int i = 0; i < 4; i++) { int j = tid + 256 * i; atomicAdd(&qss[j >> 6], qv[i] * qv[i]); }
    __syncthreads();
#pragma unroll
    for (int i = 0; i < 4; i++) {
        int j = tid + 256 * i;
        float iv = rsqrtf(qss[j >> 6] * (1.f / HD) + 1e-6f);
        qn_bf[j] = f2bf(qv[i] * iv * c_qg[j & 63]);
    }
    float oacc[4] = {0.f, 0.f, 0.f, 0.f};
    __syncthreads();

    int rbase = lane & 15, kb = (lane < 16) ? 0 : 16, koff = (lane < 16) ? 0 : 16;

    // ---- 3. stream z in 16-row tiles ----
    for (int s0 = 0; s0 < Ss; s0 += 16) {
        // (a) RMSNorm two z rows per wave, keep in regs, emit bf16 tile
#pragma unroll
        for (int rr = 0; rr < 2; rr++) {
            int r = w * 2 + rr;
            const float* zr = zbase + (size_t)(s0 + r) * Dd;
            float zv[32]; float zs = 0.f;
#pragma unroll
            for (int i = 0; i < 32; i++) { zv[i] = zr[lane + 32 * i]; zs += zv[i] * zv[i]; }
            zs = wred_sum(zs);
            float zinv = rsqrtf(zs * (1.f / Dd) + 1e-6f);
#pragma unroll
            for (int i = 0; i < 32; i++) {
                int d = lane + 32 * i;
                z_bf[r * 1032 + d] = f2bf(zv[i] * zinv * normkv_g[d]);
            }
        }
        __syncthreads();

        // (b) WMMA: [k|v](16x512) = z_tile(16x1024) @ [wk|wv]; wave covers 64 cols
        v8f acc[4];
#pragma unroll
        for (int ni = 0; ni < 4; ni++) acc[ni] = vzero8();
        int nb = w * 64;
        const ushort_t* wcolp[4];
#pragma unroll
        for (int ni = 0; ni < 4; ni++) {
            int n = nb + ni * 16 + rbase;
            wcolp[ni] = (n < 256) ? (wkT + (size_t)n * Dd) : (wvT + (size_t)(n - 256) * Dd);
        }
        for (int k0 = 0; k0 < Dd; k0 += 32) {
            FragU ua;
            const ushort_t* ar = &z_bf[rbase * 1032 + k0];
#pragma unroll
            for (int p = 0; p < 4; p++) ua.u[p]     = *(const unsigned int*)(ar + kb + 2 * p);
#pragma unroll
            for (int p = 0; p < 4; p++) ua.u[4 + p] = *(const unsigned int*)(ar + 8 + kb + 2 * p);
#pragma unroll
            for (int ni = 0; ni < 4; ni++) {
                FragU ub;
                const ushort_t* vr = wcolp[ni] + k0 + koff;
                ub.q[0] = *(const uint4*)vr;
                ub.q[1] = *(const uint4*)(vr + 8);
                acc[ni] = wmma_bf16(ua.v, ub.v, acc[ni]);
            }
        }

        // (c) k RMSNorm on accumulators (waves 0-3), v plain (waves 4-7) -> kv_bf
        if (w < 4) {
#pragma unroll
            for (int i = 0; i < 8; i++) {
                float s2 = 0.f;
#pragma unroll
                for (int ni = 0; ni < 4; ni++) s2 += acc[ni][i] * acc[ni][i];
                s2 = hred_sum(s2);
                float kinv = rsqrtf(s2 * (1.f / HD) + 1e-6f);
                int row = i + ((lane < 16) ? 0 : 8);
#pragma unroll
                for (int ni = 0; ni < 4; ni++) {
                    int col = nb + ni * 16 + (lane & 15);
                    kv_bf[row * 520 + col] = f2bf(acc[ni][i] * kinv * c_kg[col & 63]);
                }
            }
        } else {
#pragma unroll
            for (int i = 0; i < 8; i++) {
                int row = i + ((lane < 16) ? 0 : 8);
#pragma unroll
                for (int ni = 0; ni < 4; ni++) {
                    int col = nb + ni * 16 + (lane & 15);
                    kv_bf[row * 520 + col] = f2bf(acc[ni][i]);
                }
            }
        }
        __syncthreads();

        // (d) scores + softcap, thread = (head h, tile-row x16)
        const ushort_t* kr = &kv_bf[x16 * 520 + (h >> 2) * HD];
        const ushort_t* qr = &qn_bf[h * HD];
        float dot = 0.f;
#pragma unroll
        for (int d = 0; d < HD; d++) dot += bf2f(qr[d]) * bf2f(kr[d]);
        float scv = 50.f * tanhf(dot * 0.125f * (1.f / 50.f));

        // (e) online softmax, 16-lane group per head
        float mt = hred_max(scv);
        float mold = mrun[h];
        float mnew = fmaxf(mold, mt);
        float p = __expf(scv - mnew);
        float psum = hred_sum(p);
        sc_t[h * 16 + x16] = p;
        if (x16 == 0) { alpha[h] = __expf(mold - mnew); lrun[h] = lrun[h] * alpha[h] + psum; mrun[h] = mnew; }
        __syncthreads();

        // (f) o update, thread = (head h, 4 dims at x16*4)
        float al = alpha[h];
        float part[4] = {0.f, 0.f, 0.f, 0.f};
#pragma unroll
        for (int sr = 0; sr < 16; sr++) {
            float pw = sc_t[h * 16 + sr];
            const ushort_t* vr = &kv_bf[sr * 520 + 256 + (h >> 2) * HD + x16 * 4];
#pragma unroll
            for (int i = 0; i < 4; i++) part[i] += pw * bf2f(vr[i]);
        }
#pragma unroll
        for (int i = 0; i < 4; i++) oacc[i] = oacc[i] * al + part[i];
        __syncthreads();
    }

    // ---- 4. finalize o, project through c_wo, add residual ----
    float linv = 1.f / lrun[h];
#pragma unroll
    for (int i = 0; i < 4; i++) o_sh[h * HD + x16 * 4 + i] = oacc[i] * linv;
    __syncthreads();

    float outv[4] = {0.f, 0.f, 0.f, 0.f};
    for (int d = 0; d < Hh * HD; d++) {
        float ov = o_sh[d];
        const float* wr = c_wo + (size_t)d * Dd;
#pragma unroll
        for (int i = 0; i < 4; i++) outv[i] += ov * wr[tid + 256 * i];
    }
#pragma unroll
    for (int i = 0; i < 4; i++) {
        int j = tid + 256 * i;
        agent1[(size_t)bt * Dd + j] = arow[j] + outv[i];
    }
}

// ---------------------------------------------------------------------------
// Self-attention probabilities: causal + softcap + softmax. One block per (b,h)
// K-cache staged via async DMA.
// ---------------------------------------------------------------------------
__global__ __launch_bounds__(256) void k_sscores(const ushort_t* __restrict__ q_bf,
                                                 const ushort_t* __restrict__ k_bf,
                                                 ushort_t* __restrict__ attn) {
    int bh = blockIdx.x; int b = bh >> 4, h = bh & 15, kk = h >> 2;
    int lane = threadIdx.x & 31, w = threadIdx.x >> 5;
    __shared__ __align__(16) ushort_t ksh[Tt * HD];
    for (int idx = threadIdx.x; idx < (Tt * HD) / 8; idx += 256) {
        int u = idx >> 3, c = (idx & 7) * 8;
        async_copy_b128(&k_bf[((size_t)(b * Tt + u)) * (Kk * HD) + kk * HD + c],
                        &ksh[u * 64 + c]);
    }
    async_wait0();
    __syncthreads();
    for (int t = w * 32; t < w * 32 + 32; t++) {
        const ushort_t* qr = q_bf + ((size_t)(b * Tt + t)) * (Hh * HD) + h * HD;
        float sc[8];
#pragma unroll
        for (int j = 0; j < 8; j++) {
            int u = lane + 32 * j;
            float dot = 0.f;
            for (int d = 0; d < HD; d++) dot += bf2f(qr[d]) * bf2f(ksh[u * HD + d]);
            float v = 50.f * tanhf(dot * 0.125f * (1.f / 50.f));
            sc[j] = (u <= t) ? v : -1e30f;
        }
        float mx = sc[0];
#pragma unroll
        for (int j = 1; j < 8; j++) mx = fmaxf(mx, sc[j]);
        for (int m = 16; m >= 1; m >>= 1) mx = fmaxf(mx, __shfl_xor(mx, m, 32));
        float sum = 0.f;
#pragma unroll
        for (int j = 0; j < 8; j++) { sc[j] = __expf(sc[j] - mx); sum += sc[j]; }
        sum = wred_sum(sum);
        float invs = 1.f / sum;
        ushort_t* ar = attn + ((size_t)bh * Tt + t) * Tt;
#pragma unroll
        for (int j = 0; j < 8; j++) ar[lane + 32 * j] = f2bf(sc[j] * invs);
    }
}

// ---------------------------------------------------------------------------
// o = attn @ v via WMMA, one block per (b,h). V transposed in LDS -> bf16 out
// ---------------------------------------------------------------------------
__global__ __launch_bounds__(256) void k_av(const ushort_t* __restrict__ attn,
                                            const ushort_t* __restrict__ v_bf,
                                            ushort_t* __restrict__ o_bf) {
    int bh = blockIdx.x; int b = bh >> 4, h = bh & 15, kk = h >> 2;
    __shared__ __align__(16) ushort_t Vt[HD * 264];   // Vt[d][u], padded stride
    {
        // 8 chunks per thread, loads into independent regs so they pipeline
        PackU t[8];
#pragma unroll
        for (int j = 0; j < 8; j++) {
            int idx = threadIdx.x + 256 * j;
            int u = idx >> 3, db = (idx & 7) * 8;
            t[j].q = *(const uint4*)&v_bf[((size_t)(b * Tt + u)) * (Kk * HD) + kk * HD + db];
        }
#pragma unroll
        for (int j = 0; j < 8; j++) {
            int idx = threadIdx.x + 256 * j;
            int u = idx >> 3, db = (idx & 7) * 8;
#pragma unroll
            for (int e = 0; e < 8; e++) Vt[(db + e) * 264 + u] = t[j].s[e];
        }
    }
    __syncthreads();
    int lane = threadIdx.x & 31, w = threadIdx.x >> 5;
    int mb = w * 32;
    int rbase = lane & 15, kb = (lane < 16) ? 0 : 16, koff = (lane < 16) ? 0 : 16;
    v8f acc[2][4];
#pragma unroll
    for (int mi = 0; mi < 2; mi++)
#pragma unroll
        for (int ni = 0; ni < 4; ni++) acc[mi][ni] = vzero8();
    const ushort_t* abase = attn + (size_t)bh * Tt * Tt;
    for (int k0 = 0; k0 < Tt; k0 += 32) {
        FragU ua[2];
#pragma unroll
        for (int mi = 0; mi < 2; mi++) {
            const ushort_t* ar = abase + (size_t)(mb + mi * 16 + rbase) * Tt + k0;
#pragma unroll
            for (int p = 0; p < 4; p++) ua[mi].u[p]     = *(const unsigned int*)(ar + kb + 2 * p);
#pragma unroll
            for (int p = 0; p < 4; p++) ua[mi].u[4 + p] = *(const unsigned int*)(ar + 8 + kb + 2 * p);
        }
#pragma unroll
        for (int ni = 0; ni < 4; ni++) {
            int d = ni * 16 + rbase;
            FragU ub;
            const ushort_t* vr = &Vt[d * 264 + k0 + koff];
            ub.q[0] = *(const uint4*)vr;
            ub.q[1] = *(const uint4*)(vr + 8);
#pragma unroll
            for (int mi = 0; mi < 2; mi++)
                acc[mi][ni] = wmma_bf16(ua[mi].v, ub.v, acc[mi][ni]);
        }
    }
#pragma unroll
    for (int mi = 0; mi < 2; mi++)
#pragma unroll
        for (int ni = 0; ni < 4; ni++)
#pragma unroll
            for (int i = 0; i < 8; i++) {
                int row = mb + mi * 16 + i + ((lane < 16) ? 0 : 8);
                int col = ni * 16 + (lane & 15);
                o_bf[((size_t)(b * Tt + row)) * (Hh * HD) + h * HD + col] = f2bf(acc[mi][ni][i]);
            }
}

// ---------------------------------------------------------------------------
// SwiGLU elementwise: out = bf16(silu(h1) * h3)
// ---------------------------------------------------------------------------
__global__ void k_swiglu(const float* __restrict__ h1, const float* __restrict__ h3,
                         ushort_t* __restrict__ out, int n) {
    int i = blockIdx.x * blockDim.x + threadIdx.x;
    int stride = gridDim.x * blockDim.x;
    for (; i < n; i += stride) {
        float a = h1[i];
        float s = a / (1.f + __expf(-a));
        out[i] = f2bf(s * h3[i]);
    }
}

// ---------------------------------------------------------------------------
// Host orchestration
// ---------------------------------------------------------------------------
extern "C" void kernel_launch(void* const* d_in, const int* in_sizes, int n_in,
                              void* d_out, int out_size, void* d_ws, size_t ws_size,
                              hipStream_t stream) {
    (void)in_sizes; (void)n_in; (void)out_size; (void)ws_size;
    const float* agent   = (const float*)d_in[0];
    const float* z       = (const float*)d_in[1];
    const float* norm1_g = (const float*)d_in[2];
    const float* normkv_g= (const float*)d_in[3];
    const float* norm2_g = (const float*)d_in[4];
    const float* norm3_g = (const float*)d_in[5];
    const float* c_wq    = (const float*)d_in[6];
    const float* c_wk    = (const float*)d_in[7];
    const float* c_wv    = (const float*)d_in[8];
    const float* c_wo    = (const float*)d_in[9];
    const float* c_qg    = (const float*)d_in[10];
    const float* c_kg    = (const float*)d_in[11];
    const float* s_wq    = (const float*)d_in[12];
    const float* s_wk    = (const float*)d_in[13];
    const float* s_wv    = (const float*)d_in[14];
    const float* s_wo    = (const float*)d_in[15];
    const float* s_qg    = (const float*)d_in[16];
    const float* s_kg    = (const float*)d_in[17];
    const float* f_w1    = (const float*)d_in[18];
    const float* f_w3    = (const float*)d_in[19];
    const float* f_w2    = (const float*)d_in[20];

    char* ws = (char*)d_ws;
    size_t off = 0;
    auto alloc = [&](size_t bytes) -> void* {
        off = (off + 255) & ~(size_t)255;
        void* p = ws + off;
        off += bytes;
        return p;
    };
    const int BT = Bb * Tt;                 // 512 rows
    ushort_t* wkT_bf  = (ushort_t*)alloc((size_t)256 * Dd * 2);   // [n][k] bf16
    ushort_t* wvT_bf  = (ushort_t*)alloc((size_t)256 * Dd * 2);
    ushort_t* s_wq_bf = (ushort_t*)alloc((size_t)Dd * 1024 * 2);
    ushort_t* s_wk_bf = (ushort_t*)alloc((size_t)Dd * 256 * 2);
    ushort_t* s_wv_bf = (ushort_t*)alloc((size_t)Dd * 256 * 2);
    ushort_t* s_wo_bf = (ushort_t*)alloc((size_t)1024 * Dd * 2);
    ushort_t* f_w1_bf = (ushort_t*)alloc((size_t)Dd * HF * 2);
    ushort_t* f_w3_bf = (ushort_t*)alloc((size_t)Dd * HF * 2);
    ushort_t* f_w2_bf = (ushort_t*)alloc((size_t)HF * Dd * 2);
    float*    agent1  = (float*)   alloc((size_t)BT * Dd * 4);
    ushort_t* x_bf    = (ushort_t*)alloc((size_t)BT * Dd * 2);
    float*    q_s     = (float*)   alloc((size_t)BT * 1024 * 4);
    ushort_t* q_bf    = (ushort_t*)alloc((size_t)BT * 1024 * 2);
    float*    k_s     = (float*)   alloc((size_t)BT * 256 * 4);
    ushort_t* k_bf    = (ushort_t*)alloc((size_t)BT * 256 * 2);
    ushort_t* v_bf    = (ushort_t*)alloc((size_t)BT * 256 * 2);
    ushort_t* attn_bf = (ushort_t*)alloc((size_t)Bb * Hh * Tt * Tt * 2);
    ushort_t* o_bf    = (ushort_t*)alloc((size_t)BT * 1024 * 2);
    float*    agent2  = (float*)   alloc((size_t)BT * Dd * 4);
    ushort_t* y_bf    = (ushort_t*)alloc((size_t)BT * Dd * 2);
    float*    h1      = (float*)   alloc((size_t)BT * HF * 4);
    float*    h3      = (float*)   alloc((size_t)BT * HF * 4);
    ushort_t* h_bf    = (ushort_t*)alloc((size_t)BT * HF * 2);

    // ---- weight transforms (fp32 -> bf16, cross K/V also transposed) ----
    hipLaunchKernelGGL(k_transpose_bf16, dim3(256 / 32, Dd / 32), dim3(256), 0, stream,
                       c_wk, wkT_bf, Dd, 256);
    hipLaunchKernelGGL(k_transpose_bf16, dim3(256 / 32, Dd / 32), dim3(256), 0, stream,
                       c_wv, wvT_bf, Dd, 256);
    auto conv = [&](const float* s, ushort_t* d, int n) {
        int blocks = (n / 4 + 255) / 256; if (blocks > 2048) blocks = 2048;
        hipLaunchKernelGGL(k_f32_to_bf16, dim3(blocks), dim3(256), 0, stream, s, d, n);
    };
    conv(s_wq, s_wq_bf, Dd * 1024);
    conv(s_wk, s_wk_bf, Dd * 256);
    conv(s_wv, s_wv_bf, Dd * 256);
    conv(s_wo, s_wo_bf, 1024 * Dd);
    conv(f_w1, f_w1_bf, Dd * HF);
    conv(f_w3, f_w3_bf, Dd * HF);
    conv(f_w2, f_w2_bf, HF * Dd);

    // ---- fused cross-attention (z streamed once) ----
    hipLaunchKernelGGL(k_cross, dim3(BT), dim3(256), 0, stream,
                       agent, z, norm1_g, normkv_g, c_wq, wkT_bf, wvT_bf, c_wo, c_qg, c_kg, agent1);

    // ---- self-attention ----
    hipLaunchKernelGGL(k_rmsnorm_bf16, dim3(BT), dim3(256), 0, stream, agent1, norm2_g, x_bf, Dd);
    hipLaunchKernelGGL(k_gemm_bf16, dim3(1024 / 128, BT / 128), dim3(256), 0, stream,
                       x_bf, s_wq_bf, (const float*)nullptr, q_s, (ushort_t*)nullptr,
                       BT, 1024, Dd, Dd, 1024, 1024);
    hipLaunchKernelGGL(k_gemm_bf16, dim3(256 / 128, BT / 128), dim3(256), 0, stream,
                       x_bf, s_wk_bf, (const float*)nullptr, k_s, (ushort_t*)nullptr,
                       BT, 256, Dd, Dd, 256, 256);
    hipLaunchKernelGGL(k_gemm_bf16, dim3(256 / 128, BT / 128), dim3(256), 0, stream,
                       x_bf, s_wv_bf, (const float*)nullptr, (float*)nullptr, v_bf,
                       BT, 256, Dd, Dd, 256, 256);
    hipLaunchKernelGGL(k_headnorm_bf16, dim3(BT), dim3(256), 0, stream, q_s, s_qg, q_bf, 1024);
    hipLaunchKernelGGL(k_headnorm_bf16, dim3(BT), dim3(256), 0, stream, k_s, s_kg, k_bf, 256);
    hipLaunchKernelGGL(k_sscores, dim3(Bb * Hh), dim3(256), 0, stream, q_bf, k_bf, attn_bf);
    hipLaunchKernelGGL(k_av, dim3(Bb * Hh), dim3(256), 0, stream, attn_bf, v_bf, o_bf);
    hipLaunchKernelGGL(k_gemm_bf16, dim3(1024 / 128, BT / 128), dim3(256), 0, stream,
                       o_bf, s_wo_bf, agent1, agent2, (ushort_t*)nullptr,
                       BT, 1024, 1024, 1024, 1024, 1024);

    // ---- SwiGLU FFN ----
    hipLaunchKernelGGL(k_rmsnorm_bf16, dim3(BT), dim3(256), 0, stream, agent2, norm3_g, y_bf, Dd);
    hipLaunchKernelGGL(k_gemm_bf16, dim3(HF / 128, BT / 128), dim3(256), 0, stream,
                       y_bf, f_w1_bf, (const float*)nullptr, h1, (ushort_t*)nullptr,
                       BT, HF, Dd, Dd, HF, HF);
    hipLaunchKernelGGL(k_gemm_bf16, dim3(HF / 128, BT / 128), dim3(256), 0, stream,
                       y_bf, f_w3_bf, (const float*)nullptr, h3, (ushort_t*)nullptr,
                       BT, HF, Dd, Dd, HF, HF);
    {
        int n = BT * HF;
        int blocks = (n + 255) / 256; if (blocks > 2048) blocks = 2048;
        hipLaunchKernelGGL(k_swiglu, dim3(blocks), dim3(256), 0, stream, h1, h3, h_bf, n);
    }
    hipLaunchKernelGGL(k_gemm_bf16, dim3(1024 / 128, BT / 128), dim3(256), 0, stream,
                       h_bf, f_w2_bf, agent2, (float*)d_out, (ushort_t*)nullptr,
                       BT, 1024, HF, HF, 1024, 1024);
}
